// IdeficsPerceiverResampler_33904471834711
// MI455X (gfx1250) — matrix operand order repaired and checked by reading, not compiled, not run on hardware
//
#include <hip/hip_runtime.h>
#include <hip/hip_bf16.h>

typedef unsigned short u16;
typedef __attribute__((ext_vector_type(16))) __bf16 v16bf;
typedef __attribute__((ext_vector_type(8)))  float  v8f;
typedef __attribute__((ext_vector_type(4)))  unsigned int v4u;
typedef __attribute__((ext_vector_type(8)))  int v8i;
typedef __attribute__((ext_vector_type(4)))  int v4i;

#define B_    16
#define S_    4096
#define E_    1280
#define L_    64
#define HD_   1536
#define FF_   5120
#define NH_   16
#define HDIM_ 96
#define SL_   (S_ + L_)
#define DEPTH_ 3
#define KT_   (E_ / 64)   // 20 TDM tiles of 16x64 bf16 per A row-block

__device__ __forceinline__ u16 f2bf(float f) {
  unsigned u = __float_as_uint(f);
  u += 0x7FFFu + ((u >> 16) & 1u);      // round-to-nearest-even
  return (u16)(u >> 16);
}

union Frag {
  uint4   q[2];
  u16     h[16];
  v16bf   v;
};

__device__ __forceinline__ v8f zero8() {
  v8f z = {0.f, 0.f, 0.f, 0.f, 0.f, 0.f, 0.f, 0.f};
  return z;
}

// A-matrix 16x32 bf16 fragment (row-major source, per-lane row pointer).
// lanes 0-15: K = {k0+0..7, k0+16..23}; lanes 16-31: K = {k0+8..15, k0+24..31}
__device__ __forceinline__ v16bf load_a_frag(const u16* rowptr, int k0) {
  int hi = (threadIdx.x >> 4) & 1;
  Frag f;
  f.q[0] = *(const uint4*)(rowptr + k0 + hi * 8);
  f.q[1] = *(const uint4*)(rowptr + k0 + 16 + hi * 8);
  return f.v;
}

// B-matrix 32x16 bf16 fragment; B stored as [N,K] row-major (weights [out,in]),
// per-lane pointer = row n of W. lanes 0-15: K = {k0+0..7, k0+16..23}; lanes 16-31: K = {k0+8..15, k0+24..31}
__device__ __forceinline__ v16bf load_b_frag(const u16* colptr, int k0) {
  int hi = (threadIdx.x >> 4) & 1;
  Frag f;
  f.q[0] = *(const uint4*)(colptr + k0 + hi * 8);
  f.q[1] = *(const uint4*)(colptr + k0 + 16 + hi * 8);
  return f.v;
}

#define WMMA_BF16(a, b, c) \
  __builtin_amdgcn_wmma_f32_16x16x32_bf16(false, (a), false, (b), (short)0, (c), false, false)

// ------------------------------------------------------------ TDM A-tile DMA
// Stage a 16-row x 64-col bf16 tile (row stride E_ elements) from global into
// LDS at byte offset lds_off, via the Tensor Data Mover (tracked by TENSORcnt).
__device__ __forceinline__ void tdm_load_tile16x64(unsigned lds_off, const u16* gsrc) {
  unsigned long long ga = (unsigned long long)(uintptr_t)gsrc;
  v4u g0;
  g0[0] = 1u;                                            // count=1, user desc
  g0[1] = lds_off;                                       // lds_addr [63:32]
  g0[2] = (unsigned)(ga & 0xffffffffu);                  // global_addr [95:64]
  g0[3] = (unsigned)((ga >> 32) & 0x01ffffffu) | (2u << 30); // addr hi | type=2
  v8i g1;
  g1[0] = (int)(1u << 16);        // wg_mask=0, data_size=1 (2 bytes)
  g1[1] = (int)(64u << 16);       // tensor_dim0 = 64 (tile-local bounds)
  g1[2] = (int)(16u << 16);       // tensor_dim1 = 16
  g1[3] = (int)(64u << 16);       // tile_dim0 = 64
  g1[4] = 16;                     // tile_dim1 = 16, tile_dim2 = 0
  g1[5] = (int)E_;                // tensor_dim0_stride = E_ (low 32 bits)
  g1[6] = 0;                      // stride hi / dim1_stride lo
  g1[7] = 0;
  v4i gz = {0, 0, 0, 0};
#if __clang_major__ >= 23
  v8i gz8 = {0, 0, 0, 0, 0, 0, 0, 0};
  __builtin_amdgcn_tensor_load_to_lds(g0, g1, gz, gz, gz8, 0);
#else
  __builtin_amdgcn_tensor_load_to_lds(g0, g1, gz, gz, 0);
#endif
}

// ---------------------------------------------------------------- elementwise
__global__ void cvt_f32_bf16_kernel(const float* __restrict__ src,
                                    u16* __restrict__ dst, size_t n) {
  size_t i = (size_t)blockIdx.x * blockDim.x + threadIdx.x;
  size_t stride = (size_t)gridDim.x * blockDim.x;
  for (; i < n; i += stride) dst[i] = f2bf(src[i]);
}

__global__ void broadcast_lat_kernel(const float* __restrict__ lat0,
                                     float* __restrict__ lat, size_t n) {
  size_t i = (size_t)blockIdx.x * blockDim.x + threadIdx.x;
  size_t stride = (size_t)gridDim.x * blockDim.x;
  for (; i < n; i += stride) lat[i] = lat0[i % (size_t)(L_ * E_)];
}

// ------------------------------------------------------- row layernorm (E dim)
template <bool OUTBF>
__global__ void ln_rows_kernel(const float* __restrict__ x, const float* __restrict__ w,
                               const float* __restrict__ bia, void* __restrict__ out,
                               int E) {
  __shared__ float sh1[256];
  __shared__ float sh2[256];
  const int row = blockIdx.x;
  const float* xr = x + (size_t)row * E;
  float s1 = 0.f, s2 = 0.f;
  for (int i = threadIdx.x; i < E; i += blockDim.x) {
    float v = xr[i];
    s1 += v; s2 += v * v;
  }
  sh1[threadIdx.x] = s1; sh2[threadIdx.x] = s2;
  __syncthreads();
  for (int off = 128; off > 0; off >>= 1) {
    if ((int)threadIdx.x < off) {
      sh1[threadIdx.x] += sh1[threadIdx.x + off];
      sh2[threadIdx.x] += sh2[threadIdx.x + off];
    }
    __syncthreads();
  }
  const float mu = sh1[0] / (float)E;
  const float rs = rsqrtf(sh2[0] / (float)E - mu * mu + 1e-5f);
  for (int i = threadIdx.x; i < E; i += blockDim.x) {
    float v = (xr[i] - mu) * rs * w[i] + bia[i];
    if (OUTBF) ((u16*)out)[(size_t)row * E + i] = f2bf(v);
    else       ((float*)out)[(size_t)row * E + i] = v;
  }
}

// --------------------------------------------- q/k/v projection (+ head LN)
// Block = 4 waves (128 threads). All waves share one 16-row A block; the
// 16x64 bf16 A-tile is staged into LDS by the Tensor Data Mover
// (double-buffered, TENSORcnt), and each wave computes one head's
// 16x96 output tile with 6 WMMA accumulators. Optional fused per-head
// layernorm epilogue (cross-lane shfl reductions), then *scale.
__global__ void proj_ln_kernel(const u16* __restrict__ ctxn, const u16* __restrict__ latn,
                               const u16* __restrict__ W, const float* __restrict__ lnw,
                               const float* __restrict__ lnb, u16* __restrict__ out,
                               int rowsPerB, int Sctx, float scale) {
  __shared__ u16 atile[2][16 * 64];
  const int tid = threadIdx.x;
  const int lane = tid & 31, rlo = lane & 15, hi = (lane >> 4) & 1;
  const int wave = tid >> 5;                  // 0..3
  const int r0 = blockIdx.x * 16;
  const int head = blockIdx.y * 4 + wave;

  // A source: 16-row block is entirely inside one region (S_ % 16 == 0)
  const int b = r0 / rowsPerB, s0 = r0 % rowsPerB;
  const u16* abase = (s0 < Sctx) ? (ctxn + ((size_t)b * Sctx + s0) * E_)
                                 : (latn + ((size_t)b * L_ + (s0 - Sctx)) * E_);

  const u16* wrow[6];
#pragma unroll
  for (int t = 0; t < 6; t++)
    wrow[t] = W + (size_t)(head * HDIM_ + t * 16 + rlo) * E_;

  v8f acc[6];
#pragma unroll
  for (int t = 0; t < 6; t++) acc[t] = zero8();

  const unsigned lds0 = (unsigned)(uintptr_t)(&atile[0][0]);
  const unsigned lds1 = (unsigned)(uintptr_t)(&atile[1][0]);

  if (wave == 0) tdm_load_tile16x64(lds0, abase);        // prime buffer 0

  for (int kt = 0; kt < KT_; kt++) {
    const int buf = kt & 1;
    if (wave == 0) {
      if (kt + 1 < KT_) {
        tdm_load_tile16x64(buf ? lds0 : lds1, abase + (kt + 1) * 64);
        __builtin_amdgcn_s_wait_tensorcnt(1);            // tile kt complete
      } else {
        __builtin_amdgcn_s_wait_tensorcnt(0);
      }
    }
    __syncthreads();                                     // publish tile kt
#pragma unroll
    for (int sub = 0; sub < 2; sub++) {
      // A fragment straight from LDS (ds_load_b128): row rlo, 16B aligned
      Frag fa;
      const int abase_e = rlo * 64 + sub * 32 + hi * 8;
      fa.q[0] = *(const uint4*)&atile[buf][abase_e];
      fa.q[1] = *(const uint4*)&atile[buf][abase_e + 16];
      const int kg = kt * 64 + sub * 32;
      __builtin_prefetch((const void*)(wrow[0] + kg + 128), 0, 1);
#pragma unroll
      for (int t = 0; t < 6; t++)
        acc[t] = WMMA_BF16(fa.v, load_b_frag(wrow[t], kg), acc[t]);
    }
    __syncthreads();                                     // tile kt consumed
  }

  if (lnw != nullptr) {
    float sum[8], sq[8];
#pragma unroll
    for (int j = 0; j < 8; j++) { sum[j] = 0.f; sq[j] = 0.f; }
#pragma unroll
    for (int t = 0; t < 6; t++)
#pragma unroll
      for (int j = 0; j < 8; j++) {
        float v = acc[t][j];
        sum[j] += v; sq[j] += v * v;
      }
#pragma unroll
    for (int m = 1; m < 16; m <<= 1)
#pragma unroll
      for (int j = 0; j < 8; j++) {
        sum[j] += __shfl_xor(sum[j], m, 32);
        sq[j]  += __shfl_xor(sq[j],  m, 32);
      }
    float wv[6], bv[6];
#pragma unroll
    for (int t = 0; t < 6; t++) {
      wv[t] = lnw[t * 16 + rlo];
      bv[t] = lnb[t * 16 + rlo];
    }
#pragma unroll
    for (int j = 0; j < 8; j++) {
      float mu = sum[j] * (1.f / HDIM_);
      float rs = rsqrtf(sq[j] * (1.f / HDIM_) - mu * mu + 1e-5f);
      int orow = r0 + hi * 8 + j;
#pragma unroll
      for (int t = 0; t < 6; t++) {
        float v = ((acc[t][j] - mu) * rs * wv[t] + bv[t]) * scale;
        out[(size_t)orow * HD_ + head * HDIM_ + t * 16 + rlo] = f2bf(v);
      }
    }
  } else {
#pragma unroll
    for (int j = 0; j < 8; j++) {
      int orow = r0 + hi * 8 + j;
#pragma unroll
      for (int t = 0; t < 6; t++)
        out[(size_t)orow * HD_ + head * HDIM_ + t * 16 + rlo] = f2bf(acc[t][j] * scale);
    }
  }
}

// ------------------------------------------------ flash attention (one wave)
// grid.x = B*NH (b,h), grid.y = L/16 (query block). Q pre-scaled by hd^-0.5.
__global__ void attn_kernel(const u16* __restrict__ qb, const u16* __restrict__ kb,
                            const u16* __restrict__ vb, u16* __restrict__ ao) {
  __shared__ float pshm[16 * 32];
  const int lane = threadIdx.x & 31, rlo = lane & 15, hi = lane >> 4;
  const int b = blockIdx.x >> 4, h = blockIdx.x & 15;
  const int q0 = blockIdx.y * 16;

  const u16* qrow = qb + (size_t)(b * L_ + q0 + rlo) * HD_ + h * HDIM_;
  v16bf qf[3];
#pragma unroll
  for (int kk = 0; kk < 3; kk++) qf[kk] = load_a_frag(qrow, kk * 32);

  v8f o[6];
#pragma unroll
  for (int t = 0; t < 6; t++) o[t] = zero8();
  float mrow[8], lrow[8];
#pragma unroll
  for (int j = 0; j < 8; j++) { mrow[j] = -3e38f; lrow[j] = 0.f; }

  const size_t krowbase = (size_t)b * SL_ * HD_ + (size_t)h * HDIM_;

  for (int key0 = 0; key0 < SL_; key0 += 32) {
    v8f s0 = zero8(), s1 = zero8();
    const u16* kcol0 = kb + krowbase + (size_t)(key0 + rlo) * HD_;
    const u16* kcol1 = kb + krowbase + (size_t)(key0 + 16 + rlo) * HD_;
#pragma unroll
    for (int kk = 0; kk < 3; kk++) {
      s0 = WMMA_BF16(qf[kk], load_b_frag(kcol0, kk * 32), s0);
      s1 = WMMA_BF16(qf[kk], load_b_frag(kcol1, kk * 32), s1);
    }
    // online softmax over the 32 keys of this tile (rows distributed per-lane)
#pragma unroll
    for (int j = 0; j < 8; j++) {
      float tm = fmaxf(s0[j], s1[j]);
#pragma unroll
      for (int m = 1; m < 16; m <<= 1) tm = fmaxf(tm, __shfl_xor(tm, m, 32));
      float mnew  = fmaxf(mrow[j], tm);
      float alpha = __expf(mrow[j] - mnew);
      float p0 = __expf(s0[j] - mnew);
      float p1 = __expf(s1[j] - mnew);
      float ls = p0 + p1;
#pragma unroll
      for (int m = 1; m < 16; m <<= 1) ls += __shfl_xor(ls, m, 32);
      lrow[j] = lrow[j] * alpha + ls;
      mrow[j] = mnew;
#pragma unroll
      for (int t = 0; t < 6; t++) o[t][j] *= alpha;
      pshm[(hi * 8 + j) * 32 + rlo]      = p0;
      pshm[(hi * 8 + j) * 32 + 16 + rlo] = p1;
    }
    __syncthreads();
    // rebuild P as an A-fragment (16x32 bf16) from LDS
    Frag pf;
#pragma unroll
    for (int e = 0; e < 16; e++) {
      int k = hi * 16 + e;
      pf.h[e] = f2bf(pshm[rlo * 32 + k]);
    }
    // O += P @ V  (V gathered as B-fragment: rows = keys, cols = head dims)
#pragma unroll
    for (int t = 0; t < 6; t++) {
      Frag vf;
      const u16* vptr = vb + (size_t)(b * SL_ + key0 + hi * 16) * HD_ + h * HDIM_ + t * 16 + rlo;
#pragma unroll
      for (int e = 0; e < 16; e++) vf.h[e] = vptr[(size_t)e * HD_];
      o[t] = WMMA_BF16(pf.v, vf.v, o[t]);
    }
    __syncthreads();
  }

#pragma unroll
  for (int j = 0; j < 8; j++) {
    float inv = 1.f / lrow[j];
    int orow = b * L_ + q0 + hi * 8 + j;
#pragma unroll
    for (int t = 0; t < 6; t++)
      ao[(size_t)orow * HD_ + h * HDIM_ + t * 16 + rlo] = f2bf(o[t][j] * inv);
  }
}

// -------------------------------------------------- generic bf16 WMMA GEMM
// C[M,N] = A[M,K] @ W[N,K]^T (+res) (relu) ; one wave = 16x64 tile.
template <bool RELU, bool RES, bool OUTBF>
__global__ void gemm_kernel(const u16* __restrict__ A, const u16* __restrict__ W,
                            const float* __restrict__ res, void* __restrict__ out,
                            int N, int K) {
  const int lane = threadIdx.x & 31, rlo = lane & 15, hi = lane >> 4;
  const int r0 = blockIdx.x * 16, c0 = blockIdx.y * 64;
  const u16* arow = A + (size_t)(r0 + rlo) * K;
  const u16* wrow[4];
#pragma unroll
  for (int t = 0; t < 4; t++) wrow[t] = W + (size_t)(c0 + t * 16 + rlo) * K;

  v8f acc[4];
#pragma unroll
  for (int t = 0; t < 4; t++) acc[t] = zero8();

  for (int k0 = 0; k0 < K; k0 += 32) {
    v16bf a = load_a_frag(arow, k0);
    __builtin_prefetch((const void*)(arow + k0 + 128), 0, 1);
#pragma unroll
    for (int t = 0; t < 4; t++)
      acc[t] = WMMA_BF16(a, load_b_frag(wrow[t], k0), acc[t]);
  }

#pragma unroll
  for (int j = 0; j < 8; j++) {
    int row = r0 + hi * 8 + j;
#pragma unroll
    for (int t = 0; t < 4; t++) {
      int col = c0 + t * 16 + rlo;
      size_t idx = (size_t)row * N + col;
      float v = acc[t][j];
      if (RES)  v += res[idx];
      if (RELU) v = fmaxf(v, 0.f);
      if (OUTBF) ((u16*)out)[idx] = f2bf(v);
      else       ((float*)out)[idx] = v;
    }
  }
}

// ---------------------------------------------------------------------------
extern "C" void kernel_launch(void* const* d_in, const int* in_sizes, int n_in,
                              void* d_out, int out_size, void* d_ws, size_t ws_size,
                              hipStream_t stream) {
  (void)in_sizes; (void)n_in; (void)out_size; (void)ws_size;
  const float* context  = (const float*)d_in[0];
  const float* latents  = (const float*)d_in[1];
  const float* ctx_ln_w = (const float*)d_in[2];
  const float* ctx_ln_b = (const float*)d_in[3];
  const float* lat_ln_w = (const float*)d_in[4];
  const float* lat_ln_b = (const float*)d_in[5];
  const float* q_ln_w   = (const float*)d_in[6];
  const float* q_ln_b   = (const float*)d_in[7];
  const float* k_ln_w   = (const float*)d_in[8];
  const float* k_ln_b   = (const float*)d_in[9];
  const float* wq       = (const float*)d_in[10];
  const float* wk       = (const float*)d_in[11];
  const float* wv       = (const float*)d_in[12];
  const float* wo       = (const float*)d_in[13];
  const float* mlp_ln_w = (const float*)d_in[14];
  const float* mlp_ln_b = (const float*)d_in[15];
  const float* w_fc     = (const float*)d_in[16];
  const float* w_cp     = (const float*)d_in[17];
  const float* fin_w    = (const float*)d_in[18];
  const float* fin_b    = (const float*)d_in[19];

  char* p = (char*)d_ws;
  auto alloc = [&](size_t bytes) -> char* {
    char* r = p;
    p += (bytes + 255) & ~(size_t)255;
    return r;
  };
  u16*   wq_bf  = (u16*)alloc((size_t)DEPTH_ * HD_ * E_ * 2);
  u16*   wk_bf  = (u16*)alloc((size_t)DEPTH_ * HD_ * E_ * 2);
  u16*   wv_bf  = (u16*)alloc((size_t)DEPTH_ * HD_ * E_ * 2);
  u16*   wo_bf  = (u16*)alloc((size_t)DEPTH_ * E_ * HD_ * 2);
  u16*   wfc_bf = (u16*)alloc((size_t)DEPTH_ * FF_ * E_ * 2);
  u16*   wcp_bf = (u16*)alloc((size_t)DEPTH_ * E_ * FF_ * 2);
  u16*   ctxn   = (u16*)alloc((size_t)B_ * S_ * E_ * 2);
  u16*   latn   = (u16*)alloc((size_t)B_ * L_ * E_ * 2);
  u16*   qbuf   = (u16*)alloc((size_t)B_ * L_ * HD_ * 2);
  u16*   kbuf   = (u16*)alloc((size_t)B_ * SL_ * HD_ * 2);
  u16*   vbuf   = (u16*)alloc((size_t)B_ * SL_ * HD_ * 2);
  u16*   aobuf  = (u16*)alloc((size_t)B_ * L_ * HD_ * 2);
  float* latf   = (float*)alloc((size_t)B_ * L_ * E_ * 4);
  u16*   mlpn   = (u16*)alloc((size_t)B_ * L_ * E_ * 2);
  u16*   hbuf   = (u16*)alloc((size_t)B_ * L_ * FF_ * 2);

  // weight conversion (f32 -> bf16)
  cvt_f32_bf16_kernel<<<2048, 256, 0, stream>>>(wq,   wq_bf,  (size_t)DEPTH_ * HD_ * E_);
  cvt_f32_bf16_kernel<<<2048, 256, 0, stream>>>(wk,   wk_bf,  (size_t)DEPTH_ * HD_ * E_);
  cvt_f32_bf16_kernel<<<2048, 256, 0, stream>>>(wv,   wv_bf,  (size_t)DEPTH_ * HD_ * E_);
  cvt_f32_bf16_kernel<<<2048, 256, 0, stream>>>(wo,   wo_bf,  (size_t)DEPTH_ * E_ * HD_);
  cvt_f32_bf16_kernel<<<2048, 256, 0, stream>>>(w_fc, wfc_bf, (size_t)DEPTH_ * FF_ * E_);
  cvt_f32_bf16_kernel<<<2048, 256, 0, stream>>>(w_cp, wcp_bf, (size_t)DEPTH_ * E_ * FF_);

  broadcast_lat_kernel<<<1024, 256, 0, stream>>>(latents, latf, (size_t)B_ * L_ * E_);

  const float qscale = 1.0f / sqrtf((float)HDIM_);

  for (int i = 0; i < DEPTH_; i++) {
    // layernorms -> bf16
    ln_rows_kernel<true><<<B_ * S_, 256, 0, stream>>>(context, ctx_ln_w + i * E_,
                                                      ctx_ln_b + i * E_, ctxn, E_);
    ln_rows_kernel<true><<<B_ * L_, 256, 0, stream>>>(latf, lat_ln_w + i * E_,
                                                      lat_ln_b + i * E_, latn, E_);
    // projections (Q: LN + scale; K: LN; V: plain). 4 waves / block share a
    // TDM-staged LDS A-tile; each wave owns one head.
    dim3 gq(B_ * L_ / 16, NH_ / 4);
    proj_ln_kernel<<<gq, 128, 0, stream>>>(latn, latn, wq_bf + (size_t)i * HD_ * E_,
                                           q_ln_w + i * HDIM_, q_ln_b + i * HDIM_,
                                           qbuf, L_, 0, qscale);
    dim3 gkv(B_ * SL_ / 16, NH_ / 4);
    proj_ln_kernel<<<gkv, 128, 0, stream>>>(ctxn, latn, wk_bf + (size_t)i * HD_ * E_,
                                            k_ln_w + i * HDIM_, k_ln_b + i * HDIM_,
                                            kbuf, SL_, S_, 1.0f);
    proj_ln_kernel<<<gkv, 128, 0, stream>>>(ctxn, latn, wv_bf + (size_t)i * HD_ * E_,
                                            nullptr, nullptr, vbuf, SL_, S_, 1.0f);
    // flash attention
    attn_kernel<<<dim3(B_ * NH_, L_ / 16), 32, 0, stream>>>(qbuf, kbuf, vbuf, aobuf);
    // output projection + residual
    gemm_kernel<false, true, false><<<dim3(B_ * L_ / 16, E_ / 64), 32, 0, stream>>>(
        aobuf, wo_bf + (size_t)i * E_ * HD_, latf, latf, E_, HD_);
    // MLP
    ln_rows_kernel<true><<<B_ * L_, 256, 0, stream>>>(latf, mlp_ln_w + i * E_,
                                                      mlp_ln_b + i * E_, mlpn, E_);
    gemm_kernel<true, false, true><<<dim3(B_ * L_ / 16, FF_ / 64), 32, 0, stream>>>(
        mlpn, wfc_bf + (size_t)i * FF_ * E_, nullptr, hbuf, FF_, E_);
    gemm_kernel<false, true, false><<<dim3(B_ * L_ / 16, E_ / 64), 32, 0, stream>>>(
        hbuf, wcp_bf + (size_t)i * E_ * FF_, latf, latf, E_, FF_);
  }

  ln_rows_kernel<false><<<B_ * L_, 256, 0, stream>>>(latf, fin_w, fin_b, d_out, E_);
}
